// TMAN1_10479720202422
// MI455X (gfx1250) — compile-verified, matrix operand order checked by reading, not run on hardware
//
#include <hip/hip_runtime.h>
#include <hip/hip_bf16.h>

#define T_STEPS 128
#define NB      1024
#define DHID    128
#define DXL 300
#define DXA 74
#define DXV 35
#define PXL 320
#define PXA 96
#define PXV 64
#define DXT 409
#define PXT 480   // 320 + 96 + 64

typedef __attribute__((ext_vector_type(16))) __bf16 v16bf;
typedef __attribute__((ext_vector_type(8)))  float  v8f;
typedef __attribute__((ext_vector_type(4)))  float  v4f;

// bf16 weight arena layout in d_ws (element offsets)
enum : unsigned {
  OFF_WL   = 0,                      // Wih_l padded 512x320
  OFF_WA   = 163840,                 // Wih_a padded 512x96
  OFF_WV   = 212992,                 // Wih_v padded 512x64
  OFF_WHL  = 245760,                 // Whh_l 512x128
  OFF_WHA  = 311296,
  OFF_WHV  = 376832,
  OFF_ATT1 = 442368,                 // att_w1 128x384
  OFF_NN1  = 491520,                 // nn{l,a,v}_w1 3 x 128x128
  OFF_NN2  = 540672,                 // nn{l,a,v}_w2 3 x 128x128
  OFF_P1   = 589824,                 // p_w1 128x128
  OFF_P2   = 606208,                 // p_w2 128x128
  OFF_G1A  = 622592,                 // g1_w1 128x256
  OFF_G1B  = 655360,                 // g1_w2 128x128
  OFF_G2A  = 671744,                 // g2_w1 128x256
  OFF_G2B  = 704512,                 // g2_w2 128x128
  OFF_OUT1 = 720896,                 // out_w1 128x512
  WS_TOTAL = 786432                  // elements (x2 bytes = 1.5MB)
};

static __device__ __forceinline__ unsigned short f2bf(float f) {
  unsigned u = __float_as_uint(f);
  u += 0x7FFFu + ((u >> 16) & 1u);         // round-to-nearest-even
  return (unsigned short)(u >> 16);
}
static __device__ __forceinline__ float bf2f(unsigned short h) {
  return __uint_as_float(((unsigned)h) << 16);
}
static __device__ __forceinline__ float sigm(float x) {
  return 1.f / (1.f + __expf(-x));
}

union TileU { v16bf v; v4f q[2]; };

// One 16x16x32 bf16 WMMA step. a/b point at the lane's first 8 bf16 elements
// (row r = lane&15, K offset khi = (lane>=16)?8:0 already folded into a/b).
static __device__ __forceinline__ v8f wmma_bf16(const unsigned short* a,
                                                const unsigned short* b,
                                                v8f acc) {
  TileU A, B;
  A.q[0] = *(const v4f*)(a);
  A.q[1] = *(const v4f*)(a + 16);
  B.q[0] = *(const v4f*)(b);
  B.q[1] = *(const v4f*)(b + 16);
  return __builtin_amdgcn_wmma_f32_16x16x32_bf16(false, A.v, false, B.v,
                                                 (short)0, acc, false, false);
}

struct Params {
  const float* x;
  const float* bih[3]; const float* bhh[3];
  const float* att_b1; const float* att_w2; const float* att_b2;
  const float* nn_b1[3]; const float* nn_b2[3];
  const float* p_b1;  const float* p_b2;
  const float* g1_b1; const float* g1_b2;
  const float* g2_b1; const float* g2_b2;
  const float* out_b1; const float* out_w2; const float* out_b2;
  const unsigned short* w;
  float* out;
};

// fp32 -> padded bf16 weight conversion (run once, L2-resident thereafter)
__global__ void mfn_wcvt_kernel(const float* __restrict__ src,
                                unsigned short* __restrict__ dst,
                                int rows, int sc, int dc) {
  int i = blockIdx.x * 256 + threadIdx.x;
  if (i >= rows * dc) return;
  int r = i / dc, c = i - r * dc;
  float v = (c < sc) ? src[r * sc + c] : 0.f;
  dst[i] = f2bf(v);
}

// Persistent fused MFN: one block = 16 batch rows through all 128 steps.
__global__ __launch_bounds__(256, 1) void mfn_fused_kernel(Params p) {
  __shared__ float st_h[3][16 * DHID];
  __shared__ float st_c[3][16 * DHID];
  __shared__ float st_mem[16 * DHID];
  __shared__ float scratchF[16 * 512];                // gates / senti / {cHat,g1,g2}
  __shared__ alignas(16) unsigned short xtb[16 * PXT];
  __shared__ alignas(16) unsigned short sU[10240];    // hb | both+pgh | fin
  __shared__ alignas(16) unsigned short cbf[16 * 384];
  __shared__ alignas(16) unsigned short mlph[4][16 * DHID];
  __shared__ float attw[16][4];

  const int tid = threadIdx.x;
  const int wv  = tid >> 5;
  const int ln  = tid & 31;
  const int lr  = ln & 15;            // tile row (A) / tile col (B,C,D)
  const int hi8 = (ln >> 4) << 3;     // 0 or 8: K offset for A/B, row offset for C/D
  const int b0  = blockIdx.x * 16;

  for (int i = tid; i < 16 * DHID; i += 256) {
    st_h[0][i] = 0.f; st_h[1][i] = 0.f; st_h[2][i] = 0.f;
    st_c[0][i] = 0.f; st_c[1][i] = 0.f; st_c[2][i] = 0.f;
    st_mem[i] = 0.f;
  }
  __syncthreads();

  const unsigned short* Wx[3] = { p.w + OFF_WL, p.w + OFF_WA, p.w + OFF_WV };
  const int KXB[3]  = {10, 3, 2};          // K blocks of 32 for x projection
  const int XOFF[3] = {0, PXL, PXL + PXA}; // section offsets in xtb

  for (int t = 0; t < T_STEPS; ++t) {
    // ---- Phase 0: stage x_t (fp32->bf16, zero-padded sections) + h -> bf16
    const float* xt = p.x + ((size_t)t * NB + b0) * DXT;
    for (int i = tid; i < 16 * PXT; i += 256) {
      int r = i / PXT, c = i - r * PXT;
      float v = 0.f;
      if (c < PXL)            { if (c < DXL) v = xt[r * DXT + c]; }
      else if (c < PXL + PXA) { int cc = c - PXL; if (cc < DXA) v = xt[r * DXT + DXL + cc]; }
      else                    { int cc = c - PXL - PXA; if (cc < DXV) v = xt[r * DXT + DXL + DXA + cc]; }
      xtb[i] = f2bf(v);
    }
    unsigned short* hb = sU;   // [3][16*128] bf16 copy of h_{t-1}
    for (int i = tid; i < 3 * 16 * DHID; i += 256)
      hb[i] = f2bf(st_h[i >> 11][i & 2047]);
    if (t + 1 < T_STEPS) {     // prefetch next step's x slice (global_prefetch_b8)
      const float* xn = p.x + ((size_t)(t + 1) * NB + b0) * DXT;
      for (int i = tid * 16; i < 16 * DXT; i += 256 * 16)
        __builtin_prefetch(xn + i, 0, 3);
    }
    __syncthreads();

    // ---- Phase 1+2: per-modality LSTM (gate GEMM then cell update)
    for (int m = 0; m < 3; ++m) {
      const unsigned short* Wxm = Wx[m];
      const unsigned short* Whm = p.w + OFF_WHL + m * (512 * DHID);
      const float* bi = p.bih[m];
      const float* bh = p.bhh[m];
      const int kxb = KXB[m], xof = XOFF[m], ldx = KXB[m] * 32;
      for (int tt = wv; tt < 32; tt += 8) {           // 512 gate cols = 32 tiles
        const int n0 = tt * 16;
        v8f acc = {0, 0, 0, 0, 0, 0, 0, 0};
        const unsigned short* arow = xtb + lr * PXT + xof + hi8;
        const unsigned short* brow = Wxm + (n0 + lr) * ldx + hi8;
        for (int kb = 0; kb < kxb; ++kb)
          acc = wmma_bf16(arow + kb * 32, brow + kb * 32, acc);
        const unsigned short* ah  = hb + m * (16 * DHID) + lr * DHID + hi8;
        const unsigned short* bhw = Whm + (n0 + lr) * DHID + hi8;
        for (int kb = 0; kb < 4; ++kb)
          acc = wmma_bf16(ah + kb * 32, bhw + kb * 32, acc);
        const float badd = bi[n0 + lr] + bh[n0 + lr];
        #pragma unroll
        for (int i = 0; i < 8; ++i)
          scratchF[(i + hi8) * 512 + n0 + lr] = acc[i] + badd;
      }
      __syncthreads();
      for (int i = tid; i < 16 * DHID; i += 256) {
        int r = i >> 7, c = i & 127;
        float gi = scratchF[r * 512 + c];
        float gf = scratchF[r * 512 + 128 + c];
        float gg = scratchF[r * 512 + 256 + c];
        float go = scratchF[r * 512 + 384 + c];
        float cn = sigm(gf) * st_c[m][i] + sigm(gi) * tanhf(gg);
        float hn = sigm(go) * tanhf(cn);
        st_c[m][i] = cn;
        st_h[m][i] = hn;
        cbf[r * 384 + m * 128 + c] = f2bf(cn);        // new_cs (bf16) for att/nn
      }
      __syncthreads();
    }

    // ---- Phase 3: hidden layers of att (K=384) and nn_{l,a,v} (K=128): 32 tiles
    for (int tt = wv; tt < 32; tt += 8) {
      const int which = tt >> 3;                      // 0..2 = nn, 3 = att
      const int n0 = (tt & 7) * 16;
      v8f acc = {0, 0, 0, 0, 0, 0, 0, 0};
      float b1;
      if (which < 3) {
        const unsigned short* W1 = p.w + OFF_NN1 + which * (DHID * DHID);
        const unsigned short* a  = cbf + lr * 384 + which * 128 + hi8;
        const unsigned short* bp = W1 + (n0 + lr) * DHID + hi8;
        for (int kb = 0; kb < 4; ++kb) acc = wmma_bf16(a + kb * 32, bp + kb * 32, acc);
        b1 = p.nn_b1[which][n0 + lr];
      } else {
        const unsigned short* W1 = p.w + OFF_ATT1;
        const unsigned short* a  = cbf + lr * 384 + hi8;
        const unsigned short* bp = W1 + (n0 + lr) * 384 + hi8;
        for (int kb = 0; kb < 12; ++kb) acc = wmma_bf16(a + kb * 32, bp + kb * 32, acc);
        b1 = p.att_b1[n0 + lr];
      }
      #pragma unroll
      for (int i = 0; i < 8; ++i) {
        float v = acc[i] + b1;
        mlph[which][(i + hi8) * DHID + n0 + lr] = f2bf(v > 0.f ? v : 0.f);
      }
    }
    __syncthreads();

    // ---- Phase 4: senti = nn second layers (24 tiles) + att logits + softmax
    for (int tt = wv; tt < 24; tt += 8) {
      const int m = tt >> 3, n0 = (tt & 7) * 16;
      const unsigned short* W2 = p.w + OFF_NN2 + m * (DHID * DHID);
      v8f acc = {0, 0, 0, 0, 0, 0, 0, 0};
      const unsigned short* a  = mlph[m] + lr * DHID + hi8;
      const unsigned short* bp = W2 + (n0 + lr) * DHID + hi8;
      for (int kb = 0; kb < 4; ++kb) acc = wmma_bf16(a + kb * 32, bp + kb * 32, acc);
      const float b2 = p.nn_b2[m][n0 + lr];
      #pragma unroll
      for (int i = 0; i < 8; ++i)
        scratchF[m * 2048 + (i + hi8) * DHID + n0 + lr] = acc[i] + b2;   // senti
    }
    if (tid < 48) {                                    // att out-dim 3: plain dots
      const int r = tid & 15, k = tid >> 4;
      float s = p.att_b2[k];
      const unsigned short* hrow = mlph[3] + r * DHID;
      const float* w2 = p.att_w2 + k * DHID;
      for (int c = 0; c < DHID; ++c) s += bf2f(hrow[c]) * w2[c];
      attw[r][k] = s;
    }
    __syncthreads();
    if (tid < 16) {                                    // softmax over 3
      float a0 = attw[tid][0], a1 = attw[tid][1], a2 = attw[tid][2];
      float mx = fmaxf(a0, fmaxf(a1, a2));
      float e0 = __expf(a0 - mx), e1 = __expf(a1 - mx), e2 = __expf(a2 - mx);
      float inv = 1.f / (e0 + e1 + e2);
      attw[tid][0] = e0 * inv; attw[tid][1] = e1 * inv; attw[tid][2] = e2 * inv;
    }
    __syncthreads();

    // ---- Phase 5: attended + both=[attended,mem] (bf16)
    unsigned short* both = sU;              // [16][256]
    unsigned short* pgh  = sU + 16 * 256;   // [3][16*128]
    for (int i = tid; i < 16 * DHID; i += 256) {
      int r = i >> 7, c = i & 127;
      float a = attw[r][0] * scratchF[r * DHID + c]
              + attw[r][1] * scratchF[2048 + r * DHID + c]
              + attw[r][2] * scratchF[4096 + r * DHID + c];
      both[r * 256 + c]       = f2bf(a);
      both[r * 256 + 128 + c] = f2bf(st_mem[i]);
    }
    __syncthreads();

    // ---- Phase 6a: hidden of p (K=128) and g1,g2 (K=256): 24 tiles
    for (int tt = wv; tt < 24; tt += 8) {
      const int which = tt >> 3, n0 = (tt & 7) * 16;
      v8f acc = {0, 0, 0, 0, 0, 0, 0, 0};
      const unsigned short* a = both + lr * 256 + hi8;
      const unsigned short* W; int kbl; const float* b1p;
      if (which == 0)      { W = p.w + OFF_P1;  kbl = 4; b1p = p.p_b1;  }
      else if (which == 1) { W = p.w + OFF_G1A; kbl = 8; b1p = p.g1_b1; }
      else                 { W = p.w + OFF_G2A; kbl = 8; b1p = p.g2_b1; }
      const unsigned short* bp = W + (n0 + lr) * (kbl * 32) + hi8;
      for (int kb = 0; kb < kbl; ++kb) acc = wmma_bf16(a + kb * 32, bp + kb * 32, acc);
      const float b1 = b1p[n0 + lr];
      #pragma unroll
      for (int i = 0; i < 8; ++i) {
        float v = acc[i] + b1;
        pgh[which * 2048 + (i + hi8) * DHID + n0 + lr] = f2bf(v > 0.f ? v : 0.f);
      }
    }
    __syncthreads();

    // ---- Phase 6b: second layers -> cHat (tanh), gamma1/2 (sigmoid)
    for (int tt = wv; tt < 24; tt += 8) {
      const int which = tt >> 3, n0 = (tt & 7) * 16;
      const unsigned short* W = p.w + (which == 0 ? OFF_P2 : which == 1 ? OFF_G1B : OFF_G2B);
      const float* b2p = which == 0 ? p.p_b2 : which == 1 ? p.g1_b2 : p.g2_b2;
      v8f acc = {0, 0, 0, 0, 0, 0, 0, 0};
      const unsigned short* a  = pgh + which * 2048 + lr * DHID + hi8;
      const unsigned short* bp = W + (n0 + lr) * DHID + hi8;
      for (int kb = 0; kb < 4; ++kb) acc = wmma_bf16(a + kb * 32, bp + kb * 32, acc);
      const float b2 = b2p[n0 + lr];
      #pragma unroll
      for (int i = 0; i < 8; ++i) {
        float v = acc[i] + b2;
        v = (which == 0) ? tanhf(v) : sigm(v);
        scratchF[which * 2048 + (i + hi8) * DHID + n0 + lr] = v;
      }
    }
    __syncthreads();
    for (int i = tid; i < 16 * DHID; i += 256)         // mem update
      st_mem[i] = scratchF[2048 + i] * st_mem[i] + scratchF[4096 + i] * scratchF[i];
    __syncthreads();
  }

  // ---- Final: out MLP on [h_l,h_a,h_v,mem] (K=512)
  unsigned short* fin = sU;                            // [16][512] bf16
  for (int i = tid; i < 16 * 512; i += 256) {
    int r = i >> 9, c = i & 511;
    float v = (c < 384) ? st_h[c >> 7][r * DHID + (c & 127)]
                        : st_mem[r * DHID + (c & 127)];
    fin[i] = f2bf(v);
  }
  __syncthreads();
  {
    const int n0 = wv * 16;                            // 8 waves x 16 = 128 hidden
    v8f acc = {0, 0, 0, 0, 0, 0, 0, 0};
    const unsigned short* a  = fin + lr * 512 + hi8;
    const unsigned short* bp = p.w + OFF_OUT1 + (n0 + lr) * 512 + hi8;
    for (int kb = 0; kb < 16; ++kb) acc = wmma_bf16(a + kb * 32, bp + kb * 32, acc);
    const float b1 = p.out_b1[n0 + lr];
    #pragma unroll
    for (int i = 0; i < 8; ++i) {
      float v = acc[i] + b1;
      mlph[0][(i + hi8) * DHID + n0 + lr] = f2bf(v > 0.f ? v : 0.f);
    }
  }
  __syncthreads();
  if (tid < 16) {                                      // final out-dim 1
    float s = p.out_b2[0];
    const unsigned short* hr = mlph[0] + tid * DHID;
    for (int c = 0; c < DHID; ++c) s += bf2f(hr[c]) * p.out_w2[c];
    p.out[b0 + tid] = s;
  }
}

extern "C" void kernel_launch(void* const* d_in, const int* in_sizes, int n_in,
                              void* d_out, int out_size, void* d_ws, size_t ws_size,
                              hipStream_t stream) {
  (void)out_size; (void)ws_size;
  const long long XCNT = (long long)T_STEPS * NB * DXT;
  int base = 1, xi = 0;                                // dict order: x first, then params
  if (n_in > 0 && (long long)in_sizes[0] != XCNT) { base = 0; xi = n_in - 1; }
  auto F = [&](int k) -> const float* { return (const float*)d_in[base + k]; };
  unsigned short* w = (unsigned short*)d_ws;
  auto cvt = [&](const float* s, unsigned off, int rows, int sc, int dc) {
    int tot = rows * dc;
    mfn_wcvt_kernel<<<(tot + 255) / 256, 256, 0, stream>>>(s, w + off, rows, sc, dc);
  };
  cvt(F(0),  OFF_WL,  512, 300, 320);
  cvt(F(2),  OFF_WHL, 512, 128, 128);
  cvt(F(4),  OFF_WA,  512,  74,  96);
  cvt(F(6),  OFF_WHA, 512, 128, 128);
  cvt(F(8),  OFF_WV,  512,  35,  64);
  cvt(F(10), OFF_WHV, 512, 128, 128);
  cvt(F(12), OFF_ATT1, 128, 384, 384);
  cvt(F(16), OFF_NN1 + 0 * 16384, 128, 128, 128);
  cvt(F(18), OFF_NN2 + 0 * 16384, 128, 128, 128);
  cvt(F(20), OFF_NN1 + 1 * 16384, 128, 128, 128);
  cvt(F(22), OFF_NN2 + 1 * 16384, 128, 128, 128);
  cvt(F(24), OFF_NN1 + 2 * 16384, 128, 128, 128);
  cvt(F(26), OFF_NN2 + 2 * 16384, 128, 128, 128);
  cvt(F(28), OFF_P1,  128, 128, 128);
  cvt(F(30), OFF_P2,  128, 128, 128);
  cvt(F(32), OFF_G1A, 128, 256, 256);
  cvt(F(34), OFF_G1B, 128, 128, 128);
  cvt(F(36), OFF_G2A, 128, 256, 256);
  cvt(F(38), OFF_G2B, 128, 128, 128);
  cvt(F(40), OFF_OUT1, 128, 512, 512);

  Params P{};
  P.x = (const float*)d_in[xi];
  P.bih[0] = F(1);  P.bhh[0] = F(3);
  P.bih[1] = F(5);  P.bhh[1] = F(7);
  P.bih[2] = F(9);  P.bhh[2] = F(11);
  P.att_b1 = F(13); P.att_w2 = F(14); P.att_b2 = F(15);
  P.nn_b1[0] = F(17); P.nn_b2[0] = F(19);
  P.nn_b1[1] = F(21); P.nn_b2[1] = F(23);
  P.nn_b1[2] = F(25); P.nn_b2[2] = F(27);
  P.p_b1 = F(29);  P.p_b2 = F(31);
  P.g1_b1 = F(33); P.g1_b2 = F(35);
  P.g2_b1 = F(37); P.g2_b2 = F(39);
  P.out_b1 = F(41); P.out_w2 = F(42); P.out_b2 = F(43);
  P.w = w;
  P.out = (float*)d_out;
  mfn_fused_kernel<<<NB / 16, 256, 0, stream>>>(P);
}
